// Phi_85529978733124
// MI455X (gfx1250) — compile-verified, matrix operand order checked by reading, not run on hardware
//
#include <hip/hip_runtime.h>
#include <hip/hip_bf16.h>
#include <float.h>

typedef __bf16 bf16;
typedef __attribute__((ext_vector_type(16))) __bf16 v16bf;
typedef __attribute__((ext_vector_type(8)))  __bf16 v8bf;
typedef __attribute__((ext_vector_type(8)))  float  v8f;

#define DIM   128
#define HEADS 4
#define DH    32
#define EDIM  32
#define NDEMO 50000
#define NCURR 20000
#define NEDGE 800000

#define FLAG_SILU 1
#define FLAG_ACC  2

// ---------------- device helpers ----------------
__device__ __forceinline__ float silu_(float x) { return x / (1.f + __expf(-x)); }

__device__ __forceinline__ void atomicMaxF(float* addr, float val) {
  if (val >= 0.f) atomicMax((int*)addr, __float_as_int(val));
  else            atomicMin((unsigned int*)addr, __float_as_uint(val));
}

__device__ __forceinline__ void atomicAddF(float* addr, float val) {
  __hip_atomic_fetch_add(addr, val, __ATOMIC_RELAXED, __HIP_MEMORY_SCOPE_AGENT);
}

// CDNA5 async Global->LDS copy (ASYNCcnt-tracked, no VGPR round trip).
// vdst = per-lane LDS byte address, vaddr = per-lane 64-bit global address.
__device__ __forceinline__ void async_copy_b128(unsigned lds_off, const void* gptr) {
  asm volatile("global_load_async_to_lds_b128 %0, %1, off"
               :: "v"(lds_off), "v"((unsigned long long)(size_t)gptr)
               : "memory");
}
__device__ __forceinline__ void wait_async0() {
  asm volatile("s_wait_asynccnt 0" ::: "memory");
}

// ---------------- WMMA GEMM: C[M,128] = A[M,128](bf16) @ W[128,128](f32) + b ----------------
// 256 threads = 8 wave32; block computes a 128x128 tile; wave w -> rows w*16..w*16+15.
// W^T staged to LDS once (bf16); A staged in 128x32 slices, double-buffered via
// global_load_async_to_lds_b128 so the next slice streams in during the WMMA chain.
#define KSLICE 32
#define AROW   40   // A slice row stride (bf16): 80B -> rotating banks
#define BROW   136  // full-W^T row stride (bf16): 272B -> rotating banks, 16B aligned

__global__ __launch_bounds__(256) void gemm128_wmma_bf16(
    const bf16* __restrict__ A, const float* __restrict__ W,
    const float* __restrict__ bias, float* __restrict__ C,
    bf16* __restrict__ Cb, int M, int flags)
{
  __shared__ bf16 As[2][128][AROW];  // 2 x 10.24 KB
  __shared__ bf16 Bs[128][BROW];     // 34.8 KB : Bs[n][k] = W[k][n]

  const int tid  = threadIdx.x;
  const int lane = tid & 31;
  const int wave = tid >> 5;
  const int m0g  = blockIdx.x * 128;

  // ---- stage full W^T once: coalesced float4 reads, transposed bf16 stores
  for (int idx = tid; idx < DIM * (DIM / 4); idx += 256) {
    const int kk = idx >> 5, ng = idx & 31;        // k row, group of 4 n-cols
    const float4 w4 = *(const float4*)(W + (size_t)kk * DIM + ng * 4);
    Bs[ng * 4 + 0][kk] = (bf16)w4.x;
    Bs[ng * 4 + 1][kk] = (bf16)w4.y;
    Bs[ng * 4 + 2][kk] = (bf16)w4.z;
    Bs[ng * 4 + 3][kk] = (bf16)w4.w;
  }

  // ---- async-stage A slice 0 (128 rows x 32 cols = 512 x 16B chunks)
  for (int idx = tid; idx < 128 * 4; idx += 256) {
    const int r = idx >> 2, seg = idx & 3;
    int gr = m0g + r; if (gr >= M) gr = M - 1;     // clamp; guarded on store
    async_copy_b128((unsigned)(size_t)(void*)&As[0][r][seg * 8],
                    A + (size_t)gr * DIM + seg * 8);
  }
  wait_async0();
  __syncthreads();

  v8f acc[8] = {};
  const int half = (lane & 16) ? 8 : 0;
  const int mrow = (wave << 4) + (lane & 15);
  const int ncol = lane & 15;

  for (int ki = 0; ki < 4; ++ki) {
    const int buf = ki & 1;

    // prefetch next A slice into the other buffer while we compute
    if (ki < 3) {
      const int kb_next = (ki + 1) * KSLICE;
      for (int idx = tid; idx < 128 * 4; idx += 256) {
        const int r = idx >> 2, seg = idx & 3;
        int gr = m0g + r; if (gr >= M) gr = M - 1;
        async_copy_b128((unsigned)(size_t)(void*)&As[buf ^ 1][r][seg * 8],
                        A + (size_t)gr * DIM + kb_next + seg * 8);
      }
    }

    // A fragment (ISA 16-bit A layout: lanes0-15 K 0-7/16-23, lanes16-31 K 8-15/24-31)
    union { v16bf f; v8bf h[2]; } au;
    au.h[0] = *(const v8bf*)&As[buf][mrow][half];
    au.h[1] = *(const v8bf*)&As[buf][mrow][16 + half];

    // all 8 B fragments up front so the WMMAs issue back-to-back
    const int kb = ki * KSLICE;
    v16bf bfr[8];
#pragma unroll
    for (int c = 0; c < 8; ++c) {
      union { v16bf f; v8bf h[2]; } bu;
      bu.h[0] = *(const v8bf*)&Bs[(c << 4) + ncol][kb + half];
      bu.h[1] = *(const v8bf*)&Bs[(c << 4) + ncol][kb + 16 + half];
      bfr[c] = bu.f;
    }

    // same A operand on identical consecutive instructions -> reuse_a hint legal
#define WSTEP(c, RA)                                                            \
    acc[c] = __builtin_amdgcn_wmma_f32_16x16x32_bf16(                           \
        false, au.f, false, bfr[c], (short)0, acc[c], RA, false)
    WSTEP(0, true); WSTEP(1, true); WSTEP(2, true); WSTEP(3, true);
    WSTEP(4, true); WSTEP(5, true); WSTEP(6, true); WSTEP(7, false);
#undef WSTEP

    if (ki < 3) { wait_async0(); __syncthreads(); }
  }

  // epilogue: acc[c][j] -> row = m0g + wave*16 + (lane<16?0:8)+j, col = c*16 + (lane&15)
  const int rbase = m0g + (wave << 4) + ((lane & 16) ? 8 : 0);
#pragma unroll
  for (int c = 0; c < 8; ++c) {
    const int col = (c << 4) + ncol;
    const float bcol = bias ? bias[col] : 0.f;
#pragma unroll
    for (int j = 0; j < 8; ++j) {
      const int row = rbase + j;
      if (row < M) {
        float val = acc[c][j] + bcol;
        const size_t off = (size_t)row * DIM + col;
        if (flags & FLAG_ACC)  val += C[off];
        if (flags & FLAG_SILU) val = silu_(val);
        C[off] = val;
        if (Cb) Cb[off] = (bf16)val;
      }
    }
  }
}

// ---------------- f32 -> bf16 ----------------
__global__ void f32_to_bf16_kernel(const float* __restrict__ in, bf16* __restrict__ out, int n) {
  int i = blockIdx.x * 256 + threadIdx.x;
  if (i < n) out[i] = (bf16)in[i];
}

// ---------------- init segment-softmax buffers ----------------
__global__ void init_md_kernel(float* __restrict__ m, float* __restrict__ den, int n) {
  int i = blockIdx.x * 256 + threadIdx.x;
  if (i < n) { m[i] = -FLT_MAX; den[i] = 0.f; }
}

// ---------------- pass A: per-edge score + segment max ----------------
template <bool HASEDGE>
__global__ __launch_bounds__(256) void edge_score_max_kernel(
    const float* __restrict__ q, const float* __restrict__ k,
    const int* __restrict__ src, const int* __restrict__ dst,
    const float* __restrict__ attr, const float* __restrict__ We,
    float* __restrict__ score, float* __restrict__ mbuf, int E)
{
  __shared__ float sWe[HASEDGE ? EDIM * DIM : 1];
  if (HASEDGE) {
    for (int i = threadIdx.x; i < EDIM * DIM; i += 256) sWe[i] = We[i];
    __syncthreads();
  }
  const int lane = threadIdx.x & 31;
  const int e = blockIdx.x * 8 + (threadIdx.x >> 5);
  if (e >= E) return;
  const int s = src[e], d = dst[e];
  const float4 qv = *(const float4*)(q + (size_t)d * DIM + lane * 4);
  float4 kv = *(const float4*)(k + (size_t)s * DIM + lane * 4);
  if (HASEDGE) {
    float a = attr[(size_t)e * EDIM + lane];
    float ex = 0.f, ey = 0.f, ez = 0.f, ew = 0.f;
#pragma unroll
    for (int i = 0; i < EDIM; ++i) {
      float w = __shfl(a, i);
      const float4 wr = *(const float4*)&sWe[i * DIM + lane * 4];
      ex += w * wr.x; ey += w * wr.y; ez += w * wr.z; ew += w * wr.w;
    }
    kv.x += ex; kv.y += ey; kv.z += ez; kv.w += ew;
  }
  float p = qv.x * kv.x + qv.y * kv.y + qv.z * kv.z + qv.w * kv.w;
  p += __shfl_xor(p, 1);
  p += __shfl_xor(p, 2);
  p += __shfl_xor(p, 4);            // 8-lane (one head) reduction
  p *= 0.17677669529663689f;        // 1/sqrt(DH)
  const int head = lane >> 3;
  if ((lane & 7) == 0) {
    score[(size_t)e * HEADS + head] = p;
    atomicMaxF(mbuf + (size_t)d * HEADS + head, p);
  }
}

// ---------------- pass B: exp + segment sum ----------------
__global__ void edge_expsum_kernel(const float* __restrict__ score, const int* __restrict__ dst,
                                   const float* __restrict__ mbuf, float* __restrict__ den, int E)
{
  int idx = blockIdx.x * 256 + threadIdx.x;
  if (idx >= E * HEADS) return;
  int e = idx >> 2, h = idx & 3;
  int d = dst[e];
  float ex = __expf(score[idx] - mbuf[(size_t)d * HEADS + h]);
  atomicAddF(den + (size_t)d * HEADS + h, ex);
}

// ---------------- pass C: alpha * v scatter-add ----------------
template <bool HASEDGE>
__global__ __launch_bounds__(256) void edge_aggregate_kernel(
    const float* __restrict__ v, const int* __restrict__ src, const int* __restrict__ dst,
    const float* __restrict__ attr, const float* __restrict__ We,
    const float* __restrict__ score, const float* __restrict__ mbuf,
    const float* __restrict__ den, float* __restrict__ agg, int E)
{
  __shared__ float sWe[HASEDGE ? EDIM * DIM : 1];
  if (HASEDGE) {
    for (int i = threadIdx.x; i < EDIM * DIM; i += 256) sWe[i] = We[i];
    __syncthreads();
  }
  const int lane = threadIdx.x & 31;
  const int e = blockIdx.x * 8 + (threadIdx.x >> 5);
  if (e >= E) return;
  const int s = src[e], d = dst[e];
  const int head = lane >> 3;
  const float sc = score[(size_t)e * HEADS + head];
  const float mh = mbuf[(size_t)d * HEADS + head];
  const float dn = den[(size_t)d * HEADS + head];
  const float alpha = __expf(sc - mh) / (dn + 1e-16f);
  float4 vv = *(const float4*)(v + (size_t)s * DIM + lane * 4);
  if (HASEDGE) {
    float a = attr[(size_t)e * EDIM + lane];
    float ex = 0.f, ey = 0.f, ez = 0.f, ew = 0.f;
#pragma unroll
    for (int i = 0; i < EDIM; ++i) {
      float w = __shfl(a, i);
      const float4 wr = *(const float4*)&sWe[i * DIM + lane * 4];
      ex += w * wr.x; ey += w * wr.y; ez += w * wr.z; ew += w * wr.w;
    }
    vv.x += ex; vv.y += ey; vv.z += ez; vv.w += ew;
  }
  float* out = agg + (size_t)d * DIM + lane * 4;
  atomicAddF(out + 0, alpha * vv.x);
  atomicAddF(out + 1, alpha * vv.y);
  atomicAddF(out + 2, alpha * vv.z);
  atomicAddF(out + 3, alpha * vv.w);
}

// ---------------- fused LayerNorm + SiLU -> bf16 ----------------
__global__ __launch_bounds__(256) void ln_silu_kernel(
    const float* __restrict__ X, const float* __restrict__ g, const float* __restrict__ b,
    bf16* __restrict__ out, int N)
{
  const int lane = threadIdx.x & 31;
  const int row = blockIdx.x * 8 + (threadIdx.x >> 5);
  if (row >= N) return;
  const float4 x = *(const float4*)(X + (size_t)row * DIM + lane * 4);
  float s = x.x + x.y + x.z + x.w;
#pragma unroll
  for (int m = 16; m >= 1; m >>= 1) s += __shfl_xor(s, m);
  const float mu = s * (1.f / 128.f);
  float dx = x.x - mu, dy = x.y - mu, dz = x.z - mu, dw = x.w - mu;
  float ss = dx * dx + dy * dy + dz * dz + dw * dw;
#pragma unroll
  for (int m = 16; m >= 1; m >>= 1) ss += __shfl_xor(ss, m);
  const float inv = rsqrtf(ss * (1.f / 128.f) + 1e-5f);
  const int c = lane * 4;
  float y0 = silu_(dx * inv * g[c + 0] + b[c + 0]);
  float y1 = silu_(dy * inv * g[c + 1] + b[c + 1]);
  float y2 = silu_(dz * inv * g[c + 2] + b[c + 2]);
  float y3 = silu_(dw * inv * g[c + 3] + b[c + 3]);
  bf16* o = out + (size_t)row * DIM + c;
  o[0] = (bf16)y0; o[1] = (bf16)y1; o[2] = (bf16)y2; o[3] = (bf16)y3;
}

// ---------------- host-side helpers ----------------
static inline size_t align256(size_t x) { return (x + 255) & ~(size_t)255; }

static void launch_gemm(const bf16* A, const float* W, const float* b, float* C, bf16* Cb,
                        int M, int flags, hipStream_t s) {
  gemm128_wmma_bf16<<<(M + 127) / 128, 256, 0, s>>>(A, W, b, C, Cb, M, flags);
}

static void run_relation(const float* q, const float* k, const float* v,
                         const int* src, const int* dst, int ndst,
                         const float* attr, const float* We,
                         float* score, float* m, float* den, float* agg, hipStream_t s) {
  init_md_kernel<<<(ndst * HEADS + 255) / 256, 256, 0, s>>>(m, den, ndst * HEADS);
  const int eb = (NEDGE + 7) / 8;
  if (attr) {
    edge_score_max_kernel<true><<<eb, 256, 0, s>>>(q, k, src, dst, attr, We, score, m, NEDGE);
    edge_expsum_kernel<<<(NEDGE * HEADS + 255) / 256, 256, 0, s>>>(score, dst, m, den, NEDGE);
    edge_aggregate_kernel<true><<<eb, 256, 0, s>>>(v, src, dst, attr, We, score, m, den, agg, NEDGE);
  } else {
    edge_score_max_kernel<false><<<eb, 256, 0, s>>>(q, k, src, dst, nullptr, nullptr, score, m, NEDGE);
    edge_expsum_kernel<<<(NEDGE * HEADS + 255) / 256, 256, 0, s>>>(score, dst, m, den, NEDGE);
    edge_aggregate_kernel<false><<<eb, 256, 0, s>>>(v, src, dst, nullptr, nullptr, score, m, den, agg, NEDGE);
  }
}

extern "C" void kernel_launch(void* const* d_in, const int* in_sizes, int n_in,
                              void* d_out, int out_size, void* d_ws, size_t ws_size,
                              hipStream_t stream) {
  (void)in_sizes; (void)n_in; (void)out_size; (void)ws_size;
  // -------- input unpack (jax pytree order: dict keys sorted per level) --------
  const float* x_demo   = (const float*)d_in[0];
  const float* x_curr   = (const float*)d_in[1];
  const int*   src_d2c  = (const int*)d_in[2];
  const int*   dst_d2c  = (const int*)d_in[3];
  const int*   src_c2d  = (const int*)d_in[4];
  const int*   dst_c2d  = (const int*)d_in[5];
  const int*   src_tmp  = (const int*)d_in[6];
  const int*   dst_tmp  = (const int*)d_in[7];
  const float* attr_tmp = (const float*)d_in[8];
  // params: in_curr(W,b), in_demo(W,b), layers[0..1], out_curr(W,b)
  const float* in_curr_W = (const float*)d_in[9];
  const float* in_curr_b = (const float*)d_in[10];
  const float* in_demo_W = (const float*)d_in[11];
  const float* in_demo_b = (const float*)d_in[12];
  // layer block: c2d{k,q,s,v}(W,b)=8, d2c{...}=8, ln_curr(b,g), ln_demo(b,g), tmp{We,k,q,s,v}=9
  const float* out_W = (const float*)d_in[13 + 2 * 29];
  const float* out_b = (const float*)d_in[13 + 2 * 29 + 1];

  // -------- workspace carve --------
  char* ws = (char*)d_ws;
  size_t off = 0;
  auto carve = [&](size_t bytes) -> void* {
    void* p = ws + off; off += align256(bytes); return p;
  };
  bf16*  hd_bf = (bf16*)carve((size_t)NDEMO * DIM * 2);
  bf16*  hc_bf = (bf16*)carve((size_t)NCURR * DIM * 2);
  float* qbuf  = (float*)carve((size_t)NDEMO * DIM * 4);
  float* kbuf  = (float*)carve((size_t)NDEMO * DIM * 4);
  float* vbuf  = (float*)carve((size_t)NDEMO * DIM * 4);
  float* new_d = (float*)carve((size_t)NDEMO * DIM * 4);
  float* new_c = (float*)carve((size_t)NCURR * DIM * 4);
  float* score = (float*)carve((size_t)NEDGE * HEADS * 4);
  float* mseg  = (float*)carve((size_t)NDEMO * HEADS * 4);
  float* dseg  = (float*)carve((size_t)NDEMO * HEADS * 4);

  // -------- input linears (SiLU), outputs kept in bf16 only --------
  f32_to_bf16_kernel<<<(NDEMO * DIM + 255) / 256, 256, 0, stream>>>(x_demo, (bf16*)qbuf, NDEMO * DIM);
  launch_gemm((const bf16*)qbuf, in_demo_W, in_demo_b, new_d, hd_bf, NDEMO, FLAG_SILU, stream);
  f32_to_bf16_kernel<<<(NCURR * DIM + 255) / 256, 256, 0, stream>>>(x_curr, (bf16*)qbuf, NCURR * DIM);
  launch_gemm((const bf16*)qbuf, in_curr_W, in_curr_b, new_c, hc_bf, NCURR, FLAG_SILU, stream);

  // -------- layers --------
  for (int l = 0; l < 2; ++l) {
    const int B = 13 + l * 29;
    auto P = [&](int i) { return (const float*)d_in[B + i]; };
    // c2d: k(0,1) q(2,3) s(4,5) v(6,7); d2c: +8; ln_curr b=16 g=17; ln_demo b=18 g=19;
    // tmp: We=20 k(21,22) q(23,24) s(25,26) v(27,28)

    // --- relation d2c: demo -> curr ---
    launch_gemm(hc_bf, P(8 + 2), P(8 + 3), qbuf, nullptr, NCURR, 0, stream);  // q from x_dst=hc
    launch_gemm(hd_bf, P(8 + 0), P(8 + 1), kbuf, nullptr, NDEMO, 0, stream);  // k from x_src=hd
    launch_gemm(hd_bf, P(8 + 6), P(8 + 7), vbuf, nullptr, NDEMO, 0, stream);  // v
    launch_gemm(hc_bf, P(8 + 4), P(8 + 5), new_c, nullptr, NCURR, 0, stream); // skip
    run_relation(qbuf, kbuf, vbuf, src_d2c, dst_d2c, NCURR, nullptr, nullptr,
                 score, mseg, dseg, new_c, stream);

    // --- relation c2d: curr -> demo ---
    launch_gemm(hd_bf, P(2), P(3), qbuf, nullptr, NDEMO, 0, stream);          // q from hd
    launch_gemm(hc_bf, P(0), P(1), kbuf, nullptr, NCURR, 0, stream);          // k from hc
    launch_gemm(hc_bf, P(6), P(7), vbuf, nullptr, NCURR, 0, stream);          // v
    launch_gemm(hd_bf, P(4), P(5), new_d, nullptr, NDEMO, 0, stream);         // skip
    run_relation(qbuf, kbuf, vbuf, src_c2d, dst_c2d, NDEMO, nullptr, nullptr,
                 score, mseg, dseg, new_d, stream);

    // --- relation tmp: demo -> demo, with edge attributes ---
    launch_gemm(hd_bf, P(23), P(24), qbuf, nullptr, NDEMO, 0, stream);
    launch_gemm(hd_bf, P(21), P(22), kbuf, nullptr, NDEMO, 0, stream);
    launch_gemm(hd_bf, P(27), P(28), vbuf, nullptr, NDEMO, 0, stream);
    launch_gemm(hd_bf, P(25), P(26), new_d, nullptr, NDEMO, FLAG_ACC, stream); // skip (+=)
    run_relation(qbuf, kbuf, vbuf, src_tmp, dst_tmp, NDEMO, attr_tmp, P(20),
                 score, mseg, dseg, new_d, stream);

    // --- LN + SiLU -> next-layer bf16 activations ---
    ln_silu_kernel<<<(NDEMO + 7) / 8, 256, 0, stream>>>(new_d, P(19), P(18), hd_bf, NDEMO);
    ln_silu_kernel<<<(NCURR + 7) / 8, 256, 0, stream>>>(new_c, P(17), P(16), hc_bf, NCURR);
  }

  // -------- output linear --------
  launch_gemm(hc_bf, out_W, out_b, (float*)d_out, nullptr, NCURR, 0, stream);
}
